// Multihead_Self_Attention_12017318494413
// MI455X (gfx1250) — matrix operand order repaired and checked by reading, not compiled, not run on hardware
//
#include <hip/hip_runtime.h>
#include <cstdint>

typedef __attribute__((ext_vector_type(16))) _Float16 v16h;
typedef __attribute__((ext_vector_type(8)))  _Float16 v8h;
typedef __attribute__((ext_vector_type(4)))  _Float16 v4h;
typedef __attribute__((ext_vector_type(8)))  float    v8f;
typedef __attribute__((ext_vector_type(4)))  float    v4f;

#define DMODEL 1024
#define LSEQ   2048
#define NHEAD  16
#define DK     64
#define NTOK   4096   // B * L
#define NG     32     // B * NHEAD
#define NEG_INF (-__builtin_inff())

__device__ __forceinline__ v16h cat8(v8h a, v8h b) {
  return __builtin_shufflevector(a, b, 0,1,2,3,4,5,6,7,8,9,10,11,12,13,14,15);
}

__device__ __forceinline__ v8f wmma16(v16h a, v16h b, v8f c) {
  // D = A(16x32 f16) * B(32x16 f16) + C(16x16 f32)
  return __builtin_amdgcn_wmma_f32_16x16x32_f16(false, a, false, b,
                                                (short)0, c, false, false);
}

// CDNA5 async global->LDS copy (ASYNCcnt-tracked, no VGPR round trip).
// vdst = VGPR with LDS byte address, vaddr = 64-bit global address (GV mode).
__device__ __forceinline__ void async_copy_b128(uint32_t lds_addr, const void* gaddr) {
  asm volatile("global_load_async_to_lds_b128 %0, %1, off"
               :: "v"(lds_addr), "v"(gaddr) : "memory");
}
__device__ __forceinline__ void wait_async0() {
  asm volatile("s_wait_asynccnt 0x0" ::: "memory");
}

// ---------------------------------------------------------------------------
// fp32 -> fp16 conversion (vectorized 4-wide)
// ---------------------------------------------------------------------------
__global__ __launch_bounds__(256)
void cvt_f32_f16(const float* __restrict__ s, _Float16* __restrict__ d, int n) {
  int i = (blockIdx.x * 256 + threadIdx.x) * 4;
  if (i < n) {
    v4f v = *(const v4f*)(s + i);
    v4h o;
    o[0] = (_Float16)v[0]; o[1] = (_Float16)v[1];
    o[2] = (_Float16)v[2]; o[3] = (_Float16)v[3];
    *(v4h*)(d + i) = o;
  }
}

// ---------------------------------------------------------------------------
// C = A (MTx1024, f16 row-major) * W^T (W is N x K = 1024x1024 f16 row-major)
// Block: 256 thr = 8 waves; block tile 128(M) x 64(N); wave tile 16 x 64.
// W tiles double-buffered in LDS via async global->LDS copies (ASYNCcnt).
// mode 0: write f16 head-split (g,l,dk)              (Q, K)
// mode 1: write f16 head-split transposed (g,dk,l)   (V)
// mode 2: write f32 plain (tok, 1024) to d_out       (out-proj)
// ---------------------------------------------------------------------------
__global__ __launch_bounds__(256)
void gemm_nt(const _Float16* __restrict__ A, const _Float16* __restrict__ W,
             _Float16* __restrict__ outh, float* __restrict__ outf, int mode) {
  __shared__ _Float16 bsm[2][64 * 40];   // 2 x (64 W-rows x 32 k, stride 40)

  const int tid  = threadIdx.x;
  const int wave = tid >> 5;
  const int lane = tid & 31;
  const int half = lane >> 4;
  const int ln   = lane & 15;
  const int m0   = blockIdx.x * 128 + wave * 16;
  const int n0   = blockIdx.y * 64;
  const int arow = m0 + ln;
  const int klo  = half * 8;     // unused; A-frag also uses koff
  const int koff = half * 16;    // B-frag k sub-offset

  // staging assignment: each of the 256 threads owns one 16B transfer per step
  const int srow = tid >> 2;
  const int sc8  = (tid & 3) * 8;
  const _Float16* gsrc = W + (n0 + srow) * DMODEL + sc8;
  const uint32_t lds0 = (uint32_t)(uintptr_t)&bsm[0][srow * 40 + sc8];
  const uint32_t lds1 = (uint32_t)(uintptr_t)&bsm[1][srow * 40 + sc8];

  v8f acc[4] = {};

  async_copy_b128(lds0, gsrc);                     // prologue: k0 = 0 -> buf 0

  for (int k0 = 0; k0 < DMODEL; k0 += 32) {
    wait_async0();        // my copies into current buffer have landed
    __syncthreads();      // everyone's copies landed; prior-iter reads done

    if (k0 + 32 < DMODEL)                          // issue next tile into other buf
      async_copy_b128(((k0 >> 5) & 1) ? lds0 : lds1, gsrc + k0 + 32);

    const _Float16* bcur = &bsm[(k0 >> 5) & 1][0];

    // A fragment (16x32): lane = row, VGPR0-7 = k koff..koff+15
    v16h af = cat8(*(const v8h*)(A + arow * DMODEL + k0 + koff),
                   *(const v8h*)(A + arow * DMODEL + k0 + koff + 8));

#pragma unroll
    for (int t = 0; t < 4; t++) {
      const _Float16* bp = bcur + (t * 16 + ln) * 40 + koff;
      v16h bf = cat8(*(const v8h*)bp, *(const v8h*)(bp + 8));
      acc[t] = wmma16(af, bf, acc[t]);
    }
  }

#pragma unroll
  for (int t = 0; t < 4; t++) {
#pragma unroll
    for (int r = 0; r < 8; r++) {
      int m = m0 + r + half * 8;
      int e = n0 + t * 16 + ln;
      float v = acc[t][r];
      if (mode == 2) {
        outf[m * DMODEL + e] = v;
      } else {
        int b = m >> 11, l = m & 2047;
        int h = e >> 6,  dk = e & 63;
        int g = b * NHEAD + h;
        if (mode == 0) outh[(g * LSEQ + l) * DK + dk] = (_Float16)v;
        else           outh[(g * DK + dk) * LSEQ + l] = (_Float16)v;
      }
    }
  }
}

// ---------------------------------------------------------------------------
// Flash attention: one wave per (group g, 16-query tile). Online softmax.
// Q,K: (g, l, dk) f16.  V: (g, dk, l) f16 (transposed).  Ah: (tok, 1024) f16.
// ---------------------------------------------------------------------------
__global__ __launch_bounds__(256)
void attn_kernel(const _Float16* __restrict__ Qh, const _Float16* __restrict__ Kh,
                 const _Float16* __restrict__ Vt, _Float16* __restrict__ Ah) {
  __shared__ _Float16 plds[8][16 * 40];   // per-wave P staging (16 x 32, pad 40)

  const int tid  = threadIdx.x;
  const int wave = tid >> 5;
  const int lane = tid & 31;
  const int half = lane >> 4;
  const int ln   = lane & 15;
  const int task = blockIdx.x * 8 + wave;   // 4096 tasks = 32 g * 128 qtiles
  const int g    = task >> 7;
  const int q0   = (task & 127) * 16;

  const _Float16* Qg = Qh + g * LSEQ * DK;
  const _Float16* Kg = Kh + g * LSEQ * DK;
  const _Float16* Vg = Vt + g * DK * LSEQ;
  _Float16* pbuf = &plds[wave][0];

  const int qrow = q0 + ln;
  const int klo  = half * 8;
  const int koff = half * 16;

  // Q A-fragments for the two d-steps (d0 = 0, 32)
  v16h aq0 = cat8(*(const v8h*)(Qg + qrow * DK + 0 + koff),
                  *(const v8h*)(Qg + qrow * DK + 0 + koff + 8));
  v16h aq1 = cat8(*(const v8h*)(Qg + qrow * DK + 32 + koff),
                  *(const v8h*)(Qg + qrow * DK + 32 + koff + 8));

  v8f o[4] = {};
  float mrow[8], lrow[8];
#pragma unroll
  for (int r = 0; r < 8; r++) { mrow[r] = NEG_INF; lrow[r] = 0.f; }

  const int nck = (q0 + 47) >> 5;   // key chunks of 32 covering keys <= q0+15
  for (int ck = 0; ck < nck; ++ck) {
    const int kk = ck * 32;

    // ---- S = (Q K^T) * scale, two 16-key subtiles ----
    v8f s[2] = {};
#pragma unroll
    for (int t = 0; t < 2; t++) {
      const int key = kk + t * 16 + ln;       // B column n = key
      const _Float16* kp = Kg + key * DK;
      v16h bk0 = cat8(*(const v8h*)(kp + 0 + koff),  *(const v8h*)(kp + 0 + koff + 8));
      v16h bk1 = cat8(*(const v8h*)(kp + 32 + koff), *(const v8h*)(kp + 32 + koff + 8));
      s[t] = wmma16(aq0, bk0, s[t]);
      s[t] = wmma16(aq1, bk1, s[t]);
    }

    const bool domask = (kk + 31 > q0);       // only the diagonal chunk
#pragma unroll
    for (int t = 0; t < 2; t++) {
#pragma unroll
      for (int r = 0; r < 8; r++) {
        float v = s[t][r] * 0.125f;           // 1/sqrt(64)
        if (domask) {
          int key = kk + t * 16 + ln;
          int row = q0 + r + half * 8;
          if (key > row) v = NEG_INF;
        }
        s[t][r] = v;
      }
    }

    // ---- online softmax: per-row reduce across the 16 lanes of each half ----
#pragma unroll
    for (int r = 0; r < 8; r++) {
      float mx = fmaxf(s[0][r], s[1][r]);
      mx = fmaxf(mx, __shfl_xor(mx, 1));
      mx = fmaxf(mx, __shfl_xor(mx, 2));
      mx = fmaxf(mx, __shfl_xor(mx, 4));
      mx = fmaxf(mx, __shfl_xor(mx, 8));
      float mnew  = fmaxf(mrow[r], mx);
      float alpha = __expf(mrow[r] - mnew);
      mrow[r] = mnew;
      float p0 = __expf(s[0][r] - mnew);
      float p1 = __expf(s[1][r] - mnew);
      s[0][r] = p0; s[1][r] = p1;
      float ps = p0 + p1;
      ps += __shfl_xor(ps, 1);
      ps += __shfl_xor(ps, 2);
      ps += __shfl_xor(ps, 4);
      ps += __shfl_xor(ps, 8);
      lrow[r] = lrow[r] * alpha + ps;
#pragma unroll
      for (int t = 0; t < 4; t++) o[t][r] *= alpha;
    }

    // ---- P (C-layout) -> LDS -> A-layout fragment ----
#pragma unroll
    for (int t = 0; t < 2; t++)
#pragma unroll
      for (int r = 0; r < 8; r++)
        pbuf[(r + half * 8) * 40 + t * 16 + ln] = (_Float16)s[t][r];

    __builtin_amdgcn_wave_barrier();   // order ds stores before ds loads

    v16h pa = cat8(*(const v8h*)(pbuf + ln * 40 + koff),
                   *(const v8h*)(pbuf + ln * 40 + koff + 8));

    // ---- O += P * V (V transposed -> contiguous B-fragment loads) ----
#pragma unroll
    for (int t = 0; t < 4; t++) {
      const _Float16* vp = Vg + (t * 16 + ln) * LSEQ + kk + koff;
      v16h bv = cat8(*(const v8h*)vp, *(const v8h*)(vp + 8));
      o[t] = wmma16(pa, bv, o[t]);
    }
  }

  // ---- normalize and store attention output, merged-head layout ----
  const int b = g >> 4, h = g & 15;
#pragma unroll
  for (int t = 0; t < 4; t++) {
#pragma unroll
    for (int r = 0; r < 8; r++) {
      int row = q0 + r + half * 8;
      int e   = h * DK + t * 16 + ln;
      int tok = b * LSEQ + row;
      Ah[tok * DMODEL + e] = (_Float16)(o[t][r] / lrow[r]);
    }
  }
}

// ---------------------------------------------------------------------------
extern "C" void kernel_launch(void* const* d_in, const int* in_sizes, int n_in,
                              void* d_out, int out_size, void* d_ws, size_t ws_size,
                              hipStream_t stream) {
  (void)in_sizes; (void)n_in; (void)out_size; (void)ws_size;
  const float* x  = (const float*)d_in[0];
  const float* wq = (const float*)d_in[1];
  const float* wk = (const float*)d_in[2];
  const float* wv = (const float*)d_in[3];
  const float* wo = (const float*)d_in[4];
  float* out = (float*)d_out;

  _Float16* ws  = (_Float16*)d_ws;
  _Float16* Xh  = ws;                                   // 4096x1024
  _Float16* Wqh = Xh  + (size_t)NTOK * DMODEL;          // 1024x1024 each
  _Float16* Wkh = Wqh + (size_t)DMODEL * DMODEL;
  _Float16* Wvh = Wkh + (size_t)DMODEL * DMODEL;
  _Float16* Woh = Wvh + (size_t)DMODEL * DMODEL;
  _Float16* Qh  = Woh + (size_t)DMODEL * DMODEL;        // (32,2048,64)
  _Float16* Kh  = Qh  + (size_t)NG * LSEQ * DK;
  _Float16* Vt  = Kh  + (size_t)NG * LSEQ * DK;         // (32,64,2048)
  _Float16* Ah  = Vt  + (size_t)NG * DK * LSEQ;         // (4096,1024)

  const int nX = NTOK * DMODEL, nW = DMODEL * DMODEL;
  cvt_f32_f16<<<(nX / 4 + 255) / 256, 256, 0, stream>>>(x,  Xh,  nX);
  cvt_f32_f16<<<(nW / 4 + 255) / 256, 256, 0, stream>>>(wq, Wqh, nW);
  cvt_f32_f16<<<(nW / 4 + 255) / 256, 256, 0, stream>>>(wk, Wkh, nW);
  cvt_f32_f16<<<(nW / 4 + 255) / 256, 256, 0, stream>>>(wv, Wvh, nW);
  cvt_f32_f16<<<(nW / 4 + 255) / 256, 256, 0, stream>>>(wo, Woh, nW);

  dim3 gg(NTOK / 128, DMODEL / 64);   // 32 x 16
  gemm_nt<<<gg, 256, 0, stream>>>(Xh, Wqh, Qh, nullptr, 0);
  gemm_nt<<<gg, 256, 0, stream>>>(Xh, Wkh, Kh, nullptr, 0);
  gemm_nt<<<gg, 256, 0, stream>>>(Xh, Wvh, Vt, nullptr, 1);

  attn_kernel<<<(NG * (LSEQ / 16)) / 8, 256, 0, stream>>>(Qh, Kh, Vt, Ah);

  gemm_nt<<<gg, 256, 0, stream>>>(Ah, Woh, nullptr, out, 2);
}
